// PKNeuralODE_65652870087146
// MI455X (gfx1250) — compile-verified, matrix operand order checked by reading, not run on hardware
//
#include <hip/hip_runtime.h>

// PK Neural-ODE: batch 65536, state 2, MLP 2->64->64->2 (ReLU), RK4 fixed-step,
// dense output at 64 linspace points per segment, bolus doses at 24/48/72h.
// 64x64 layer on v_wmma_f32_16x16x32_f16 (f32 accumulate). Layer 1 is computed
// per-lane directly into WMMA A-fragment registers (no LDS round trip); W1/b1
// and all W2 B-fragments are register-resident for the whole integration.

typedef __attribute__((ext_vector_type(16))) _Float16 v16h;
typedef __attribute__((ext_vector_type(8)))  float    v8f;

#define HID  64
#define SPB  64    // samples per block
#define BLK  128   // 4 wave32
#define NSEG 4

struct Smem {
  _Float16 w2[HID][HID];   // W2[k][n] f16 (B-operand source, init only)
  float    h2[SPB][HID];   // layer-2 activations (C/D scatter -> layer 3)
  float    ys[SPB][2];     // stage input states (broadcast)
  float    w1[2][HID];     // init staging
  float    b1v[HID];       // init staging
  float    b2v[HID];
  float    w3[HID][2];
  float    b3v[2];
};

// One evaluation of dy = MLP(y) for the block's 64 samples. All 128 threads
// participate (WMMA needs full EXEC); only tid < SPB own a sample / get dy.
// Exactly 2 barriers per eval; the caller's next eval's first barrier orders
// layer-3 h2 reads against the next eval's h2 writes.
__device__ __forceinline__ void dyn_eval(
    Smem* s, const v16h* bf, const float* b2n,
    const v16h& w1x0, const v16h& w1x1, const v16h& w1y0, const v16h& w1y1,
    const v16h& b1f0, const v16h& b1f1,
    int tid, int lane, int wave,
    float yi0, float yi1, float* dy0, float* dy1) {
  if (tid < SPB) { s->ys[tid][0] = yi0; s->ys[tid][1] = yi1; }
  __syncthreads();

  // ---- layer 1 (2 -> 64): packed f16 FMA straight into A fragments ----
  const int srow = (wave << 4) + (lane & 15);
  const _Float16 y0h = (_Float16)s->ys[srow][0];
  const _Float16 y1h = (_Float16)s->ys[srow][1];
  const _Float16 zh  = (_Float16)0.0f;
  v16h a0, a1;                      // K = 0..31 and K = 32..63 (A layout)
#pragma unroll
  for (int e = 0; e < 16; ++e) {
    _Float16 v0 = y0h * w1x0[e] + y1h * w1y0[e] + b1f0[e];
    _Float16 v1 = y0h * w1x1[e] + y1h * w1y1[e] + b1f1[e];
    a0[e] = (v0 > zh) ? v0 : zh;
    a1[e] = (v1 > zh) ? v1 : zh;
  }

  // ---- layer 2 (64 -> 64): 8x WMMA, bias preloaded in C layout ----
  const int m0 = (wave << 4) + ((lane >> 4) << 3);  // C/D: M = vgpr + 8*(lane>=16)
  const int nn = lane & 15;
#pragma unroll
  for (int t = 0; t < 4; ++t) {                     // 4 N-tiles of 16
    v8f c;
#pragma unroll
    for (int r = 0; r < 8; ++r) c[r] = b2n[t];
    c = __builtin_amdgcn_wmma_f32_16x16x32_f16(false, a0, false, bf[2 * t + 0],
                                               (short)0, c, false, false);
    c = __builtin_amdgcn_wmma_f32_16x16x32_f16(false, a1, false, bf[2 * t + 1],
                                               (short)0, c, false, false);
    int n = (t << 4) + nn;
#pragma unroll
    for (int r = 0; r < 8; ++r) s->h2[m0 + r][n] = fmaxf(c[r], 0.0f);
  }
  __syncthreads();

  // ---- layer 3 (64 -> 2): VALU f32 by owner threads ----
  if (tid < SPB) {
    float d0 = s->b3v[0], d1 = s->b3v[1];
#pragma unroll 8
    for (int n = 0; n < HID; ++n) {
      float h = s->h2[tid][n];
      d0 = fmaf(h, s->w3[n][0], d0);
      d1 = fmaf(h, s->w3[n][1], d1);
    }
    *dy0 = d0; *dy1 = d1;
  }
  // no trailing barrier needed (next eval's first barrier provides ordering)
}

__global__ __launch_bounds__(BLK) void pk_node_kernel(
    const float* __restrict__ y0g, const float* __restrict__ W1,
    const float* __restrict__ b1,  const float* __restrict__ W2,
    const float* __restrict__ b2,  const float* __restrict__ W3,
    const float* __restrict__ b3,  const int* __restrict__ tfp,
    const int* __restrict__ spsp,  float* __restrict__ out) {
  __shared__ Smem s;
  const int tid  = threadIdx.x;
  const int lane = tid & 31;
  const int wave = tid >> 5;

  // cooperative load of weights; W2 converted to f16 once
  for (int i = tid; i < HID * HID; i += BLK) s.w2[i >> 6][i & 63] = (_Float16)W2[i];
  for (int i = tid; i < 2 * HID;   i += BLK) s.w1[i >> 6][i & 63] = W1[i];
  for (int i = tid; i < HID;       i += BLK) { s.b1v[i] = b1[i]; s.b2v[i] = b2[i]; }
  for (int i = tid; i < HID * 2;   i += BLK) s.w3[i >> 1][i & 1] = W3[i];
  if (tid < 2) s.b3v[tid] = b3[tid];
  __syncthreads();

  const int kb = (lane < 16) ? 0 : 8;  // 16-bit A/B layout K-offset

  // W2 B-operand fragments (32x16 f16 layout), register-resident forever.
  v16h bf[8];   // [n_tile*2 + k_chunk]
  {
    const int nn = lane & 15;
#pragma unroll
    for (int t = 0; t < 4; ++t) {
#pragma unroll
      for (int kc = 0; kc < 2; ++kc) {
        const int kbase = (kc << 5) + kb;
        v16h f;
#pragma unroll
        for (int e = 0; e < 8; ++e) {
          f[e]     = s.w2[kbase + e][(t << 4) + nn];
          f[8 + e] = s.w2[kbase + 16 + e][(t << 4) + nn];
        }
        bf[2 * t + kc] = f;
      }
    }
  }

  // Per-lane W1/b1 in A-fragment element order, register-resident forever.
  v16h w1x0, w1x1, w1y0, w1y1, b1f0, b1f1;
#pragma unroll
  for (int e = 0; e < 8; ++e) {
    const int k0 = kb + e, k1 = 16 + kb + e, k2 = 32 + kb + e, k3 = 48 + kb + e;
    w1x0[e] = (_Float16)s.w1[0][k0]; w1x0[8 + e] = (_Float16)s.w1[0][k1];
    w1x1[e] = (_Float16)s.w1[0][k2]; w1x1[8 + e] = (_Float16)s.w1[0][k3];
    w1y0[e] = (_Float16)s.w1[1][k0]; w1y0[8 + e] = (_Float16)s.w1[1][k1];
    w1y1[e] = (_Float16)s.w1[1][k2]; w1y1[8 + e] = (_Float16)s.w1[1][k3];
    b1f0[e] = (_Float16)s.b1v[k0];   b1f0[8 + e] = (_Float16)s.b1v[k1];
    b1f1[e] = (_Float16)s.b1v[k2];   b1f1[8 + e] = (_Float16)s.b1v[k3];
  }
  float b2n[4];
#pragma unroll
  for (int t = 0; t < 4; ++t) b2n[t] = s.b2v[(t << 4) + (lane & 15)];

  const int   steps = spsp[0];               // 64
  const float tf    = (float)tfp[0];         // 96
  const float segt[NSEG + 1] = {0.0f, 24.0f, 48.0f, 72.0f, tf};

  const int owner = blockIdx.x * SPB + tid;
  float y0v = 0.0f, y1v = 0.0f;
  if (tid < SPB) { y0v = y0g[2 * owner]; y1v = y0g[2 * owner + 1]; }

#define DYN(YI0, YI1, K0, K1) \
  dyn_eval(&s, bf, b2n, w1x0, w1x1, w1y0, w1y1, b1f0, b1f1, \
           tid, lane, wave, (YI0), (YI1), (K0), (K1))

  for (int seg = 0; seg < NSEG; ++seg) {
    const float hstep = (segt[seg + 1] - segt[seg]) / (float)(steps - 1);
    if (tid < SPB) {  // first eval point == (post-dose) segment-start state
      int o = ((owner * NSEG + seg) * steps) * 2;
      out[o] = y0v; out[o + 1] = y1v;
    }
    for (int st = 1; st < steps; ++st) {
      float k10 = 0, k11 = 0, k20 = 0, k21 = 0;
      float k30 = 0, k31 = 0, k40 = 0, k41 = 0;
      DYN(y0v, y1v, &k10, &k11);
      DYN(y0v + 0.5f * hstep * k10, y1v + 0.5f * hstep * k11, &k20, &k21);
      DYN(y0v + 0.5f * hstep * k20, y1v + 0.5f * hstep * k21, &k30, &k31);
      DYN(y0v + hstep * k30, y1v + hstep * k31, &k40, &k41);
      y0v += hstep * (k10 + 2.0f * k20 + 2.0f * k30 + k40) * (1.0f / 6.0f);
      y1v += hstep * (k11 + 2.0f * k21 + 2.0f * k31 + k41) * (1.0f / 6.0f);
      if (tid < SPB) {
        int o = ((owner * NSEG + seg) * steps + st) * 2;
        out[o] = y0v; out[o + 1] = y1v;
      }
    }
    if (seg < NSEG - 1 && tid < SPB) y0v += 100.0f;  // bolus into compartment 0
  }
#undef DYN
}

extern "C" void kernel_launch(void* const* d_in, const int* in_sizes, int n_in,
                              void* d_out, int out_size, void* d_ws, size_t ws_size,
                              hipStream_t stream) {
  (void)n_in; (void)out_size; (void)d_ws; (void)ws_size;
  const float* y0 = (const float*)d_in[0];
  const float* W1 = (const float*)d_in[1];
  const float* b1 = (const float*)d_in[2];
  const float* W2 = (const float*)d_in[3];
  const float* b2 = (const float*)d_in[4];
  const float* W3 = (const float*)d_in[5];
  const float* b3 = (const float*)d_in[6];
  const int*   tf = (const int*)d_in[7];
  const int*   sp = (const int*)d_in[8];
  float* out = (float*)d_out;

  const int batch  = in_sizes[0] / 2;     // 65536
  const int blocks = batch / SPB;         // 1024

  pk_node_kernel<<<dim3(blocks), dim3(BLK), 0, stream>>>(
      y0, W1, b1, W2, b2, W3, b3, tf, sp, out);
}
